// LTBinaryClassifier_25271587570083
// MI455X (gfx1250) — compile-verified, matrix operand order checked by reading, not run on hardware
//
#include <hip/hip_runtime.h>
#include <hip/hip_bf16.h>
#include <math.h>

typedef __attribute__((ext_vector_type(16))) __bf16 v16bf;
typedef __attribute__((ext_vector_type(8)))  __bf16 v8bf;
typedef __attribute__((ext_vector_type(8)))  float  v8f;
typedef __attribute__((ext_vector_type(4)))  unsigned int v4u;
typedef __attribute__((ext_vector_type(8)))  int v8i;
typedef __attribute__((ext_vector_type(4)))  int v4i;

#define DIM          256     // IN_SIZE + bias
#define NSPLIT       1023
#define NROWS_A      1024    // A padded to 1024 rows (row 1023 zeroed) in workspace
#define ROWS_PER_WG  64
#define THREADS      128
#define XA_STRIDE    2050    // bf16 units/row: [0,1024)=XA cols, [1024,2047)=q, pad -> bank-friendly
#define XS_STRIDE    264     // x tile row stride (padded, 16B-multiple)
#define AS_STRIDE    264     // A block row stride: 256 + 8 pad (matches TDM pad config)

#if defined(__gfx1250__) && __has_builtin(__builtin_amdgcn_tensor_load_to_lds)
#define USE_TDM 1
#else
#define USE_TDM 0
#endif

// ---- prep: convert A (f32, 1023x256) -> bf16 (1024x256) in workspace, row 1023 = 0 ----
__global__ __launch_bounds__(256)
void cvtA_kernel(const float* __restrict__ A, __bf16* __restrict__ dst)
{
    int i = blockIdx.x * 256 + threadIdx.x;          // 0 .. 1024*256-1
    float v = (i < NSPLIT * DIM) ? A[i] : 0.0f;
    dst[i] = (__bf16)v;
}

__global__ __launch_bounds__(THREADS)
void lt_fused_kernel(const float* __restrict__ X,
                     const __bf16* __restrict__ Abf,   // bf16 A in workspace
                     const float* __restrict__ W,
                     const float* __restrict__ bscal,
                     float* __restrict__ out)
{
    extern __shared__ __align__(16) char smem[];
    __bf16* xaq = (__bf16*)smem;                          // [64][XA_STRIDE]
    __bf16* xs  = xaq + ROWS_PER_WG * XA_STRIDE;          // [64][XS_STRIDE]
    __bf16* as0 = xs  + ROWS_PER_WG * XS_STRIDE;          // [16][AS_STRIDE] ping
    __bf16* as1 = as0 + 16 * AS_STRIDE;                   // [16][AS_STRIDE] pong

    const int tid     = threadIdx.x;
    const int wg      = blockIdx.x;
    const int lane    = tid & 31;
    const int wid     = tid >> 5;
    const int rowbase = wg * ROWS_PER_WG;

    // ---- stage x tile (append bias column of 1.0) as bf16, coalesced ----
    #pragma unroll 4
    for (int j = 0; j < (ROWS_PER_WG * DIM) / THREADS; ++j) {
        int idx = tid + j * THREADS;
        int r = idx >> 8, k = idx & 255;
        float v = (k < 255) ? X[(size_t)(rowbase + r) * 255 + k] : 1.0f;
        xs[r * XS_STRIDE + k] = (__bf16)v;
    }

#if USE_TDM
    // Loop-invariant D# group1: workgroup_mask=0, data_size=1(2B), pad_enable,
    // pad_interval=6 (128 DW = one 512B row), pad_amount=3 (4 DW = 16B)  -> row stride 528B,
    // tensor_dim0=256, tensor_dim1=1024, tile_dim0=256, tile_dim1=16, dim0_stride=256.
    v8i g1;
    g1[0] = (int)((1u << 16) | (1u << 20) | (6u << 22) | (3u << 25));
    g1[1] = (int)(256u << 16);        // tensor_dim0 low16 (bits 63:48 of group)
    g1[2] = (int)(1024u << 16);       // tensor_dim0 hi16=0 | tensor_dim1 low16
    g1[3] = (int)(256u << 16);        // tensor_dim1 hi16=0 | tile_dim0
    g1[4] = 16;                       // tile_dim1 | tile_dim2=0
    g1[5] = 256;                      // tensor_dim0_stride low32
    g1[6] = 0;
    g1[7] = 0;
    v4i gz4 = {};                     // groups 2/3 unused (2D tensor)
    v8i gz8 = {};                     // unused extension group (6-arg builtin form)
    const unsigned lds_as0 = (unsigned)(unsigned long long)(uintptr_t)as0;
    const unsigned lds_as1 = (unsigned)(unsigned long long)(uintptr_t)as1;
    const unsigned long long gaBase = (unsigned long long)(uintptr_t)Abf;

    if (wid == 0) {                   // prologue: DMA block 0 into as0
        v4u g0;
        g0[0] = 1u;                   // count=1 valid descriptor
        g0[1] = lds_as0;
        g0[2] = (unsigned)gaBase;
        g0[3] = (unsigned)(gaBase >> 32) | (2u << 30);   // addr[56:32] | type=2
        __builtin_amdgcn_tensor_load_to_lds(g0, g1, gz4, gz4, gz8, 0);
    }
#else
    // fallback prologue: synchronous bf16 staging of block 0
    #pragma unroll
    for (int j = 0; j < 4; ++j) {
        int idx = tid + j * THREADS;            // 0..511 : 16 rows x 32 x (8 bf16)
        int n = idx >> 5, k8 = idx & 31;
        uint4 t = ((const uint4*)Abf)[idx];
        *(uint4*)&as0[n * AS_STRIDE + 8 * k8] = t;
    }
#endif
    __syncthreads();

    // ---- build persistent A-operand fragments (16 x-rows per wave, K=256 -> 8 frags) ----
    // 16-bit A-matrix 16x32 layout: lane (m = lane&15, hi = lane>>4),
    // element j holds K = kb*32 + hi*8 + (j<8 ? j : j+8)  -> two contiguous 8-elem runs.
    const int m  = lane & 15;
    const int hi = lane >> 4;
    const int xr = wid * 16 + m;
    v16bf afrag[8];
    #pragma unroll
    for (int kb = 0; kb < 8; ++kb) {
        const __bf16* p0 = &xs[xr * XS_STRIDE + kb * 32 + hi * 8];
        v8bf lo = *(const v8bf*)p0;          // ds_load_b128
        v8bf h8 = *(const v8bf*)(p0 + 16);   // ds_load_b128
        afrag[kb] = __builtin_shufflevector(lo, h8,
            0, 1, 2, 3, 4, 5, 6, 7, 8, 9, 10, 11, 12, 13, 14, 15);
    }

    const int asoff = m * AS_STRIDE + hi * 8;

    // ---- GEMM: XA = x @ A^T, 64 blocks of 16 nodes; TDM-pipelined A staging ----
    for (int tb = 0; tb < 64; ++tb) {
#if USE_TDM
        if (wid == 0) __builtin_amdgcn_s_wait_tensorcnt(0);  // cur block landed
#endif
        __syncthreads();   // all see cur ready; prev buffer's readers finished
        const __bf16* cur = (tb & 1) ? as1 : as0;
        const bool have_next = (tb + 1) < 64;

#if USE_TDM
        if (wid == 0 && have_next) {   // DMA next block into the other buffer
            unsigned long long ga = gaBase + (unsigned long long)(tb + 1) * (16 * DIM * 2);
            v4u g0;
            g0[0] = 1u;
            g0[1] = (tb & 1) ? lds_as0 : lds_as1;
            g0[2] = (unsigned)ga;
            g0[3] = (unsigned)(ga >> 32) | (2u << 30);
            __builtin_amdgcn_tensor_load_to_lds(g0, g1, gz4, gz4, gz8, 0);
        }
#else
        uint4 tmp[4];
        if (have_next) {
            const uint4* src = (const uint4*)(Abf + (size_t)(tb + 1) * 16 * DIM);
            #pragma unroll
            for (int j = 0; j < 4; ++j) tmp[j] = src[tid + j * THREADS];
        }
#endif

        // load ALL b-fragments for this block, then run two independent WMMA chains
        v8bf blo[8], bhi8[8];
        #pragma unroll
        for (int kb = 0; kb < 8; ++kb) {
            const __bf16* q0 = cur + asoff + kb * 32;
            blo[kb]  = *(const v8bf*)q0;
            bhi8[kb] = *(const v8bf*)(q0 + 16);
        }
        v8f c0 = {}, c1 = {};
        #pragma unroll
        for (int kb = 0; kb < 8; kb += 2) {
            v16bf bf0 = __builtin_shufflevector(blo[kb], bhi8[kb],
                0, 1, 2, 3, 4, 5, 6, 7, 8, 9, 10, 11, 12, 13, 14, 15);
            c0 = __builtin_amdgcn_wmma_f32_16x16x32_bf16(
                     false, afrag[kb], false, bf0, (short)0, c0, false, false);
            v16bf bf1 = __builtin_shufflevector(blo[kb + 1], bhi8[kb + 1],
                0, 1, 2, 3, 4, 5, 6, 7, 8, 9, 10, 11, 12, 13, 14, 15);
            c1 = __builtin_amdgcn_wmma_f32_16x16x32_bf16(
                     false, afrag[kb + 1], false, bf1, (short)0, c1, false, false);
        }
        v8f c = c0 + c1;

        // scatter C (lane holds col n=m, rows v+8*hi) as bf16 into XA
        const int col = tb * 16 + m;
        #pragma unroll
        for (int v = 0; v < 8; ++v) {
            int rr = wid * 16 + v + 8 * hi;
            xaq[rr * XA_STRIDE + col] = (__bf16)c[v];
        }

#if !USE_TDM
        if (have_next) {
            __bf16* nxt = (tb & 1) ? as0 : as1;
            #pragma unroll
            for (int j = 0; j < 4; ++j) {
                int idx = tid + j * THREADS;
                int n = idx >> 5, k8 = idx & 31;
                *(uint4*)&nxt[n * AS_STRIDE + 8 * k8] = tmp[j];
            }
        }
#endif
    }
    __syncthreads();

    // ---- tree pass: 2 lanes per row (even/odd children), no barriers needed ----
    {
        const int   row  = tid >> 1;           // 0..63
        const int   half = tid & 1;            // 0 -> even child (+), 1 -> odd child (-)
        const float sgn  = half ? -1.0f : 1.0f;
        __bf16* xarow = &xaq[row * XA_STRIDE];
        __bf16* qrow  = xarow + 1024;          // q_t stored at qrow[t], t in [0,1023)

        // bias b + root node contribution (q=1 -> clip=1 -> W_z[0]=W[256])
        float acc = (half == 0) ? (bscal[0] + W[256]) : 0.0f;
        // x . W[0:256]
        for (int j = 0; j < 128; ++j) {
            int k = 2 * j + half;
            acc = fmaf((float)xs[row * XS_STRIDE + k], W[k], acc);
        }
        #pragma unroll
        for (int l = 1; l <= 10; ++l) {
            const int base  = (1 << l) - 1;
            const int pbase = (1 << (l - 1)) - 1;
            const int cnt2  = 1 << (l - 1);
            for (int j = 0; j < cnt2; ++j) {
                int t = base + 2 * j + half;       // node id at level l
                int p = pbase + j;                 // parent split-node id
                float qp = (l == 1) ? 1.0f : (float)qrow[p];
                float xa = (float)xarow[p];
                float qv = fminf(qp, sgn * xa);
                if (l < 10) qrow[t] = (__bf16)qv;  // leaves consumed immediately
                float qc = fminf(fmaxf(qv, 0.0f), 1.0f);
                acc = fmaf(qc, W[256 + t], acc);
            }
        }
        float tot = acc + __shfl_xor(acc, 1, 32);  // combine even/odd lane pair
        if (half == 0)
            out[rowbase + row] = 1.0f / (1.0f + expf(-tot));
    }
}

extern "C" void kernel_launch(void* const* d_in, const int* in_sizes, int n_in,
                              void* d_out, int out_size, void* d_ws, size_t ws_size,
                              hipStream_t stream) {
    (void)in_sizes; (void)n_in; (void)ws_size; (void)out_size;
    const float* X = (const float*)d_in[0];
    const float* A = (const float*)d_in[1];
    const float* W = (const float*)d_in[2];
    const float* b = (const float*)d_in[3];
    float* out = (float*)d_out;
    __bf16* Abf = (__bf16*)d_ws;                 // 1024*256 bf16 = 512 KB scratch

    cvtA_kernel<<<(NROWS_A * DIM) / 256, 256, 0, stream>>>(A, Abf);

    const size_t smem_bytes =
        (size_t)(ROWS_PER_WG * XA_STRIDE + ROWS_PER_WG * XS_STRIDE + 2 * 16 * AS_STRIDE)
        * sizeof(__bf16);  // ~313 KB, fits 320 KB/WGP

    (void)hipFuncSetAttribute((const void*)lt_fused_kernel,
                              hipFuncAttributeMaxDynamicSharedMemorySize,
                              (int)smem_bytes);

    lt_fused_kernel<<<32768 / ROWS_PER_WG, THREADS, smem_bytes, stream>>>(X, Abf, W, b, out);
}
